// BiLSTM_CRF_bak_50508815401169
// MI455X (gfx1250) — matrix operand order repaired and checked by reading, not compiled, not run on hardware
//
#include <hip/hip_runtime.h>

typedef __bf16 bf16;
typedef __attribute__((ext_vector_type(8)))  __bf16 v8bf;
typedef __attribute__((ext_vector_type(16))) __bf16 v16bf;
typedef __attribute__((ext_vector_type(8)))  float  v8f;

#define BB 32
#define TT 512
#define EE 512
#define HH 512
#define GG 2048
#define H2 1024
#define LL 32
#define BT (BB*TT)

// ---------------- fragment helpers ----------------

__device__ __forceinline__ v16bf make_frag(v8bf lo, v8bf hi) {
  v16bf r;
#pragma unroll
  for (int i = 0; i < 8; ++i) { r[i] = lo[i]; r[i + 8] = hi[i]; }
  return r;
}

// A-matrix 16x32 bf16 fragment: lane<16 holds row=lane, K={k..k+7, k+16..k+23};
// lane>=16 holds row=lane-16, K={k+8..k+15, k+24..k+31}.
__device__ __forceinline__ v16bf load_fragA(const bf16* base, int ld, int lane, int kk) {
  const int row  = lane & 15;
  const int half = lane >> 4;
  const bf16* p = base + (size_t)row * ld + kk + half * 8;
  return make_frag(*(const v8bf*)p, *(const v8bf*)(p + 16));
}

// B-matrix 32x16 bf16 fragment: lane holds column nrow, K={k+8h..k+8h+7, k+16+8h..k+23+8h}
// (h = lane half, same order as A). nrow already includes (lane&15).
__device__ __forceinline__ v16bf load_fragB(const bf16* base, int ld, int nrow, int lane, int kk) {
  const int half = lane >> 4;
  const bf16* p = base + (size_t)nrow * ld + kk + half * 8;
  return make_frag(*(const v8bf*)p, *(const v8bf*)(p + 16));
}

__device__ __forceinline__ v16bf load_fragA_lds(const bf16* a_lds, int ld, int lcol, int lhalf, int kk) {
  const bf16* ap = a_lds + lcol * ld + kk + lhalf * 8;
  return make_frag(*(const v8bf*)ap, *(const v8bf*)(ap + 16));
}

__device__ __forceinline__ v8f wmma_bf16(v16bf a, v16bf b, v8f c) {
  return __builtin_amdgcn_wmma_f32_16x16x32_bf16(false, a, false, b, (short)0, c, false, false);
}

__device__ __forceinline__ float sigmoidf_(float x) { return 1.f / (1.f + __expf(-x)); }

// ---------------- conversion / gather kernels ----------------

__global__ void cast_bf16_kernel(bf16* __restrict__ dst, const float* __restrict__ src, int n) {
  int i = blockIdx.x * blockDim.x + threadIdx.x;
  if (i < n) dst[i] = (bf16)src[i];
}

__global__ void embed_kernel(bf16* __restrict__ xbf, const float* __restrict__ emb,
                             const int* __restrict__ ids) {
  size_t stride = (size_t)gridDim.x * blockDim.x;
  for (size_t i = blockIdx.x * (size_t)blockDim.x + threadIdx.x; i < (size_t)BT * EE; i += stride) {
    int bt = (int)(i >> 9);      // /512
    int e  = (int)(i & 511);
    xbf[i] = (bf16)emb[(size_t)ids[bt] * EE + e];
  }
}

// ---------------- xg = x @ W_ih^T + b_ih + b_hh  (bf16 WMMA, f32 out) --------
// C is BT x 2048. Block = 8 waves sharing one 16-row A tile (staged in LDS);
// wave computes a 16(M) x 64(N) tile; K = 512. B (weights, L2-resident) is
// ping-pong double-buffered (unroll-by-2, no register copies) so each WMMA
// group waits only on loads issued one K-step earlier.

__global__ __launch_bounds__(256) void xg_gemm_kernel(
    float* __restrict__ xg, const bf16* __restrict__ xbf, const bf16* __restrict__ Wih,
    const float* __restrict__ bih, const float* __restrict__ bhh) {
  const int wave  = threadIdx.x >> 5;
  const int lane  = threadIdx.x & 31;
  const int lcol  = lane & 15;
  const int lhalf = lane >> 4;
  const int mtile  = blockIdx.x >> 2;                 // 1024 M-tiles
  const int nchunk = (blockIdx.x & 3) * 8 + wave;     // 32 N-chunks of 64
  const int mbase = mtile * 16;
  const int nb    = nchunk * 64;

  // stage the block-shared 16x512 A tile in LDS (16 KB)
  __shared__ __align__(16) bf16 a_lds[16 * EE];
  {
    const uint4* s = (const uint4*)(xbf + (size_t)mbase * EE);
    uint4* d = (uint4*)a_lds;
    for (int i = threadIdx.x; i < 16 * EE / 8; i += 256) d[i] = s[i];
  }
  __syncthreads();

  v8f acc[4];
#pragma unroll
  for (int j = 0; j < 4; ++j) {
    float bv = bih[nb + j * 16 + lcol] + bhh[nb + j * 16 + lcol];
#pragma unroll
    for (int r = 0; r < 8; ++r) acc[j][r] = bv;
  }

  v16bf bA[4], bB[4];
#pragma unroll
  for (int j = 0; j < 4; ++j) bA[j] = load_fragB(Wih, EE, nb + j * 16 + lcol, lane, 0);

  int kk = 0;
  for (; kk < EE - 64; kk += 64) {
#pragma unroll
    for (int j = 0; j < 4; ++j) bB[j] = load_fragB(Wih, EE, nb + j * 16 + lcol, lane, kk + 32);
    __builtin_prefetch(Wih + (size_t)(nb + lcol) * EE + kk + 96, 0, 1);  // global_prefetch_b8
    {
      v16bf a = load_fragA_lds(a_lds, EE, lcol, lhalf, kk);
#pragma unroll
      for (int j = 0; j < 4; ++j) acc[j] = wmma_bf16(a, bA[j], acc[j]);
    }
#pragma unroll
    for (int j = 0; j < 4; ++j) bA[j] = load_fragB(Wih, EE, nb + j * 16 + lcol, lane, kk + 64);
    {
      v16bf a = load_fragA_lds(a_lds, EE, lcol, lhalf, kk + 32);
#pragma unroll
      for (int j = 0; j < 4; ++j) acc[j] = wmma_bf16(a, bB[j], acc[j]);
    }
  }
  // peeled tail: kk == EE-64, two K-steps remain
#pragma unroll
  for (int j = 0; j < 4; ++j) bB[j] = load_fragB(Wih, EE, nb + j * 16 + lcol, lane, kk + 32);
  {
    v16bf a = load_fragA_lds(a_lds, EE, lcol, lhalf, kk);
#pragma unroll
    for (int j = 0; j < 4; ++j) acc[j] = wmma_bf16(a, bA[j], acc[j]);
  }
  {
    v16bf a = load_fragA_lds(a_lds, EE, lcol, lhalf, kk + 32);
#pragma unroll
    for (int j = 0; j < 4; ++j) acc[j] = wmma_bf16(a, bB[j], acc[j]);
  }

#pragma unroll
  for (int j = 0; j < 4; ++j)
#pragma unroll
    for (int r = 0; r < 8; ++r)
      xg[(size_t)(mbase + r + lhalf * 8) * GG + nb + j * 16 + lcol] = acc[j][r];
}

// ---------------- LSTM recurrence ----------------
// grid = (2 batch-chunks, 2 directions), block = 1024 (32 waves on one WGP).
// h (16x512 bf16) lives in LDS; c lives in accumulator-layout registers.
// Wave w owns h-columns [w*16, w*16+16): computes gate tiles i/f/g/o at
// n = g*512 + w*16, so the cell update is wave-local. W_hh latency is hidden
// by 8 waves/SIMD + zero-register global_prefetch of the next K-slice.

__global__ __launch_bounds__(1024, 1) void lstm_kernel(
    const float* __restrict__ xg_f, const float* __restrict__ xg_b,
    const bf16* __restrict__ Whh_f, const bf16* __restrict__ Whh_b,
    bf16* __restrict__ hs) {
  const int dir = blockIdx.y;
  const int b0  = blockIdx.x * 16;
  const float* xg  = dir ? xg_b  : xg_f;
  const bf16*  Whh = dir ? Whh_b : Whh_f;

  __shared__ __align__(16) bf16 h_lds[16 * 512];

  const int lane  = threadIdx.x & 31;
  const int wave  = threadIdx.x >> 5;
  const int col0  = wave * 16;
  const int lcol  = lane & 15;
  const int lhalf = lane >> 4;

  for (int i = threadIdx.x; i < 16 * 512; i += blockDim.x) h_lds[i] = (bf16)0.f;
  v8f c_st;
#pragma unroll
  for (int r = 0; r < 8; ++r) c_st[r] = 0.f;
  __syncthreads();

  for (int s = 0; s < TT; ++s) {
    const int t = dir ? (TT - 1 - s) : s;

    // seed accumulators with xg (= x@W_ih^T + biases), matching C layout
    v8f acc[4];
#pragma unroll
    for (int g = 0; g < 4; ++g)
#pragma unroll
      for (int r = 0; r < 8; ++r) {
        int brow = b0 + lhalf * 8 + r;
        acc[g][r] = xg[((size_t)brow * TT + t) * GG + g * HH + col0 + lcol];
      }

    // h @ W_hh^T over K=512
#pragma unroll 4
    for (int kk = 0; kk < HH; kk += 32) {
      v16bf a = load_fragA_lds(h_lds, 512, lcol, lhalf, kk);   // LDS A fragment
#pragma unroll
      for (int g = 0; g < 4; ++g) {
        v16bf bf = load_fragB(Whh, HH, g * HH + col0 + lcol, lane, kk);
        acc[g] = wmma_bf16(a, bf, acc[g]);
        if (kk + 32 < HH)
          __builtin_prefetch(Whh + (size_t)(g * HH + col0 + lcol) * HH + kk + 32, 0, 1);
      }
    }

    __syncthreads();   // everyone done reading previous h
#pragma unroll
    for (int r = 0; r < 8; ++r) {
      float iv = sigmoidf_(acc[0][r]);
      float fv = sigmoidf_(acc[1][r]);
      float gv = tanhf(acc[2][r]);
      float ov = sigmoidf_(acc[3][r]);
      float cv = fv * c_st[r] + iv * gv;
      c_st[r] = cv;
      float hv = ov * tanhf(cv);
      int rrow = lhalf * 8 + r;
      h_lds[rrow * 512 + col0 + lcol] = (bf16)hv;
      hs[((size_t)(b0 + rrow) * TT + t) * H2 + dir * HH + col0 + lcol] = (bf16)hv;
    }
    __syncthreads();   // new h visible before next step's WMMA reads
  }
}

// ---------------- emissions = h @ fc_w^T + fc_b ----------------
// M = BT (one 16-row tile per wave), N = 32 (2 tiles), K = 1024.
// The whole 32x1024 bf16 fc_w (64 KB) is staged in LDS per block; the streamed
// A operand (h) is ping-pong double-buffered from global.

__global__ __launch_bounds__(256) void fc_kernel(
    float* __restrict__ em, const bf16* __restrict__ hsb,
    const bf16* __restrict__ fcw, const float* __restrict__ fcb) {
  const int wave  = threadIdx.x >> 5;
  const int lane  = threadIdx.x & 31;
  const int lcol  = lane & 15;
  const int lhalf = lane >> 4;
  const int mbase = (blockIdx.x * 8 + wave) * 16;

  __shared__ __align__(16) bf16 w_lds[LL * H2];   // 64 KB
  {
    const uint4* s = (const uint4*)fcw;
    uint4* d = (uint4*)w_lds;
    for (int i = threadIdx.x; i < LL * H2 / 8; i += 256) d[i] = s[i];
  }
  __syncthreads();

  v8f acc[2];
#pragma unroll
  for (int j = 0; j < 2; ++j) {
    float bv = fcb[j * 16 + lcol];
#pragma unroll
    for (int r = 0; r < 8; ++r) acc[j][r] = bv;
  }

  const bf16* Arow = hsb + (size_t)mbase * H2;
  v16bf aA = load_fragA(Arow, H2, lane, 0);
  v16bf aB;
  int kk = 0;
  for (; kk < H2 - 64; kk += 64) {
    aB = load_fragA(Arow, H2, lane, kk + 32);
#pragma unroll
    for (int j = 0; j < 2; ++j) {
      v16bf bf = load_fragB(w_lds, H2, j * 16 + lcol, lane, kk);
      acc[j] = wmma_bf16(aA, bf, acc[j]);
    }
    aA = load_fragA(Arow, H2, lane, kk + 64);
#pragma unroll
    for (int j = 0; j < 2; ++j) {
      v16bf bf = load_fragB(w_lds, H2, j * 16 + lcol, lane, kk + 32);
      acc[j] = wmma_bf16(aB, bf, acc[j]);
    }
  }
  // peeled tail: two K-steps remain
  aB = load_fragA(Arow, H2, lane, kk + 32);
#pragma unroll
  for (int j = 0; j < 2; ++j) {
    v16bf bf = load_fragB(w_lds, H2, j * 16 + lcol, lane, kk);
    acc[j] = wmma_bf16(aA, bf, acc[j]);
  }
#pragma unroll
  for (int j = 0; j < 2; ++j) {
    v16bf bf = load_fragB(w_lds, H2, j * 16 + lcol, lane, kk + 32);
    acc[j] = wmma_bf16(aB, bf, acc[j]);
  }

#pragma unroll
  for (int j = 0; j < 2; ++j)
#pragma unroll
    for (int r = 0; r < 8; ++r)
      em[(size_t)(mbase + r + lhalf * 8) * LL + j * 16 + lcol] = acc[j][r];
}

// ---------------- CRF NLL (one wave per batch row) ----------------

__global__ void crf_kernel(float* __restrict__ llh, const float* __restrict__ em,
                           const int* __restrict__ tags, const unsigned char* __restrict__ mask,
                           const float* __restrict__ start_t, const float* __restrict__ end_t,
                           const float* __restrict__ trans) {
  const int b = blockIdx.x;
  const int j = threadIdx.x;   // 32 lanes == 32 labels
  __shared__ float tr[LL * LL];
  for (int i = j; i < LL * LL; i += 32) tr[i] = trans[i];
  __syncthreads();

  const float* emb = em + (size_t)b * TT * LL;
  float alpha = start_t[j] + emb[j];
  for (int t = 1; t < TT; ++t) {
    float m = -INFINITY, ssum = 0.f;
#pragma unroll 8
    for (int i = 0; i < LL; ++i) {
      float ai = __shfl(alpha, i, 32);
      float v = ai + tr[i * LL + j];
      if (v > m) { ssum = ssum * __expf(m - v) + 1.f; m = v; }
      else       { ssum += __expf(v - m); }
    }
    float nxt = m + __logf(ssum) + emb[t * LL + j];
    float mk  = (float)mask[b * TT + t];
    alpha = (mk > 0.f) ? nxt : alpha;
  }
  // partition = logsumexp_j(alpha + end_t)
  float v = alpha + end_t[j];
  float m = v;
#pragma unroll
  for (int o = 16; o > 0; o >>= 1) m = fmaxf(m, __shfl_xor(m, o, 32));
  float e = __expf(v - m);
#pragma unroll
  for (int o = 16; o > 0; o >>= 1) e += __shfl_xor(e, o, 32);
  float partition = m + __logf(e);

  // numerator (lane-parallel over t)
  const int* tg = tags + (size_t)b * TT;
  float num = 0.f;
  for (int t = 1 + j; t < TT; t += 32) {
    float mk = (float)mask[b * TT + t];
    num += mk * (tr[tg[t - 1] * LL + tg[t]] + emb[t * LL + tg[t]]);
  }
#pragma unroll
  for (int o = 16; o > 0; o >>= 1) num += __shfl_xor(num, o, 32);
  int cnt = 0;
  for (int t = j; t < TT; t += 32) cnt += mask[b * TT + t] ? 1 : 0;
#pragma unroll
  for (int o = 16; o > 0; o >>= 1) cnt += __shfl_xor(cnt, o, 32);

  if (j == 0) {
    int seq_end = cnt - 1;
    float score0 = start_t[tg[0]] + emb[tg[0]];
    float numer = score0 + num + end_t[tg[seq_end]];
    llh[b] = numer - partition;
  }
}

__global__ void final_kernel(float* __restrict__ out, const float* __restrict__ llh) {
  float v = llh[threadIdx.x];
#pragma unroll
  for (int o = 16; o > 0; o >>= 1) v += __shfl_xor(v, o, 32);
  if (threadIdx.x == 0) out[0] = -v / (float)BB;
}

// ---------------- host launcher ----------------

static size_t align256(size_t x) { return (x + 255) & ~(size_t)255; }

extern "C" void kernel_launch(void* const* d_in, const int* in_sizes, int n_in,
                              void* d_out, int out_size, void* d_ws, size_t ws_size,
                              hipStream_t stream) {
  (void)in_sizes; (void)n_in; (void)out_size; (void)ws_size;
  const int*   input_ids = (const int*)d_in[0];
  const int*   labels    = (const int*)d_in[1];
  const unsigned char* mask = (const unsigned char*)d_in[2];
  const float* embedding = (const float*)d_in[3];
  const float* W_ih_f = (const float*)d_in[4];
  const float* W_hh_f = (const float*)d_in[5];
  const float* b_ih_f = (const float*)d_in[6];
  const float* b_hh_f = (const float*)d_in[7];
  const float* W_ih_b = (const float*)d_in[8];
  const float* W_hh_b = (const float*)d_in[9];
  const float* b_ih_b = (const float*)d_in[10];
  const float* b_hh_b = (const float*)d_in[11];
  const float* fc_w   = (const float*)d_in[12];
  const float* fc_b   = (const float*)d_in[13];
  const float* start_t = (const float*)d_in[14];
  const float* end_t   = (const float*)d_in[15];
  const float* trans   = (const float*)d_in[16];

  char* w = (char*)d_ws;
  size_t off = 0;
  auto take = [&](size_t bytes) { char* p = w + off; off += align256(bytes); return p; };

  bf16*  xbf     = (bf16*) take((size_t)BT * EE * 2);
  bf16*  wihf_bf = (bf16*) take((size_t)GG * EE * 2);
  bf16*  wihb_bf = (bf16*) take((size_t)GG * EE * 2);
  bf16*  whhf_bf = (bf16*) take((size_t)GG * HH * 2);
  bf16*  whhb_bf = (bf16*) take((size_t)GG * HH * 2);
  bf16*  fcw_bf  = (bf16*) take((size_t)LL * H2 * 2);
  float* xg_f    = (float*)take((size_t)BT * GG * 4);
  float* xg_b    = (float*)take((size_t)BT * GG * 4);
  bf16*  hsbuf   = (bf16*) take((size_t)BT * H2 * 2);
  float* emis    = (float*)take((size_t)BT * LL * 4);
  float* llh     = (float*)take((size_t)BB * 4);

  // weight casts to bf16
  cast_bf16_kernel<<<(GG * EE + 255) / 256, 256, 0, stream>>>(wihf_bf, W_ih_f, GG * EE);
  cast_bf16_kernel<<<(GG * HH + 255) / 256, 256, 0, stream>>>(whhf_bf, W_hh_f, GG * HH);
  cast_bf16_kernel<<<(GG * EE + 255) / 256, 256, 0, stream>>>(wihb_bf, W_ih_b, GG * EE);
  cast_bf16_kernel<<<(GG * HH + 255) / 256, 256, 0, stream>>>(whhb_bf, W_hh_b, GG * HH);
  cast_bf16_kernel<<<(LL * H2 + 255) / 256, 256, 0, stream>>>(fcw_bf, fc_w, LL * H2);

  // embedding gather + cast
  embed_kernel<<<8192, 256, 0, stream>>>(xbf, embedding, input_ids);

  // input-gate precompute GEMMs: 32768 waves each (1024 Mtiles x 32 Nchunks)
  xg_gemm_kernel<<<4096, 256, 0, stream>>>(xg_f, xbf, wihf_bf, b_ih_f, b_hh_f);
  xg_gemm_kernel<<<4096, 256, 0, stream>>>(xg_b, xbf, wihb_bf, b_ih_b, b_hh_b);

  // recurrence: (2 batch chunks) x (2 directions)
  lstm_kernel<<<dim3(2, 2), 1024, 0, stream>>>(xg_f, xg_b, whhf_bf, whhb_bf, hsbuf);

  // emissions
  fc_kernel<<<128, 256, 0, stream>>>(emis, hsbuf, fcw_bf, fc_b);

  // CRF NLL
  crf_kernel<<<BB, 32, 0, stream>>>(llh, emis, labels, mask, start_t, end_t, trans);
  final_kernel<<<1, 32, 0, stream>>>((float*)d_out, llh);
}
